// QBasicBlock_ins_56753697849949
// MI455X (gfx1250) — 16-bit and 8-bit matrix operand order repaired and checked by reading, compiled, not run on hardware
//
#include <hip/hip_runtime.h>

typedef __attribute__((ext_vector_type(16))) _Float16 v16h;
typedef __attribute__((ext_vector_type(8)))  _Float16 v8h;
typedef __attribute__((ext_vector_type(8)))  float    v8f;
typedef __attribute__((ext_vector_type(8)))  int      v8i;
typedef __attribute__((__vector_size__(16))) int      v4i;

#if __has_builtin(__builtin_amdgcn_global_load_async_to_lds_b128) && \
    __has_builtin(__builtin_amdgcn_s_wait_asynccnt)
#define USE_ASYNC 1
#else
#define USE_ASYNC 0
#endif

#if USE_ASYNC
__device__ __forceinline__ __attribute__((address_space(1))) v4i* to_glb(const void* p) {
  return (__attribute__((address_space(1))) v4i*)(uintptr_t)p;
}
__device__ __forceinline__ __attribute__((address_space(3))) v4i* to_lds(void* p) {
  return (__attribute__((address_space(3))) v4i*)(unsigned)(uintptr_t)p;
}
#define ASYNC_B128(src, dst) __builtin_amdgcn_global_load_async_to_lds_b128(to_glb(src), to_lds(dst), 0, 0)
#define WAIT_ASYNC() __builtin_amdgcn_s_wait_asynccnt(0)
#else
#define ASYNC_B128(src, dst) (*(v4i*)(dst) = *(const v4i*)(src))
#define WAIT_ASYNC() ((void)0)
#endif


__global__ void zero_u32_k(unsigned* p, int n) {
  int i = blockIdx.x * blockDim.x + threadIdx.x;
  if (i < n) p[i] = 0u;
}

__global__ void maxabs_k(const float* __restrict__ w, int n, unsigned* __restrict__ out) {
  float m = 0.f;
  for (int i = blockIdx.x * blockDim.x + threadIdx.x; i < n; i += gridDim.x * blockDim.x)
    m = fmaxf(m, fabsf(w[i]));
  #pragma unroll
  for (int off = 16; off; off >>= 1) m = fmaxf(m, __shfl_xor(m, off, 32));
  if ((threadIdx.x & 31) == 0) atomicMax(out, __float_as_uint(m));
}

__global__ void cvt_f16_k(const float* __restrict__ in, _Float16* __restrict__ o, int n) {
  for (int i = blockIdx.x * blockDim.x + threadIdx.x; i < n; i += gridDim.x * blockDim.x)
    o[i] = (_Float16)in[i];
}

template<int CIN, int R>
__global__ void pack_w_k(const float* __restrict__ w, const unsigned* __restrict__ mb,
                         _Float16* __restrict__ wq, int M) {
  const int K = R * R * CIN;
  const float mx = __uint_as_float(*mb);
  const float scale = mx * (1.f / 127.f);
  const float inv = 127.f / mx;
  const int total = M * K;
  for (int i = blockIdx.x * blockDim.x + threadIdx.x; i < total; i += gridDim.x * blockDim.x) {
    int m = i / K, k = i - m * K;
    int r = k / CIN, c = k - r * CIN;
    float v = w[((size_t)m * CIN + c) * (R * R) + r];
    float q = fminf(fmaxf(rintf(v * inv), -127.f), 127.f) * scale;
    wq[i] = (_Float16)q;
  }
}

template<int CIN, int R>
__global__ void pack_w8_k(const float* __restrict__ w, const unsigned* __restrict__ mb,
                          signed char* __restrict__ wq, int M) {
  const int K = R * R * CIN;
  const float inv = 127.f / __uint_as_float(*mb);
  const int total = M * K;
  for (int i = blockIdx.x * blockDim.x + threadIdx.x; i < total; i += gridDim.x * blockDim.x) {
    int m = i / K, k = i - m * K;
    int r = k / CIN, c = k - r * CIN;
    float v = w[((size_t)m * CIN + c) * (R * R) + r];
    wq[i] = (signed char)(int)fminf(fmaxf(rintf(v * inv), -127.f), 127.f);
  }
}


template<int CIN, int R, int STRIDE, int PAD, int HIN>
__global__ __launch_bounds__(256, 2) void conv_wmma_k(
    const _Float16* __restrict__ W,
    const _Float16* __restrict__ act,
    float* __restrict__ out)
{
  constexpr int K  = R * R * CIN;
  constexpr int HW = HIN * HIN;
  constexpr int NK = K / 32;

  __shared__ __align__(16) _Float16 As[2][128][32];
  __shared__ __align__(16) _Float16 Bs[2][128][32];

  const int tid  = threadIdx.x;
  const int lane = tid & 31;
  const int wv   = tid >> 5;
  const int wm   = (wv >> 1) * 32;
  const int wn   = (wv & 1) * 64;
  const int mBlk = blockIdx.y * 128;
  const int nBlk = blockIdx.x * 128;

  const int aRow  = tid >> 1;
  const int aCol  = (tid & 1) * 16;
  const int bN    = tid >> 1;
  const int bK16  = (tid & 1) * 16;

  const int gp   = nBlk + bN;
  const int nImg = gp / 784;
  const int pix  = gp - nImg * 784;
  const int oy   = pix / 28;
  const int ox   = pix - oy * 28;
  const int iy0  = oy * STRIDE - PAD;
  const int ix0  = ox * STRIDE - PAD;

  const _Float16* aSrc = W + (size_t)(mBlk + aRow) * K + aCol;
  const _Float16* actN = act + (size_t)nImg * CIN * HW;

  v8f acc[2][4] = {};

  auto stageA = [&](int ks, int b) {
    const _Float16* src = aSrc + (size_t)ks * 32;
    _Float16* dst = &As[b][aRow][aCol];
    ASYNC_B128(src, dst);
    ASYNC_B128(src + 8, dst + 8);
  };

  auto stageB = [&](int ks, int b) {
    const int k0 = ks * 32 + bK16;
    const int r  = k0 / CIN;
    const int c0 = k0 - r * CIN;
    const int kh = r / R, kw = r - kh * R;
    const int iy = iy0 + kh, ix = ix0 + kw;
    const bool inb = (iy >= 0) && (iy < HIN) && (ix >= 0) && (ix < HIN);
    const _Float16* p = actN + ((size_t)c0 * HIN + iy) * HIN + ix;
    _Float16 tmp[16];
    #pragma unroll
    for (int j = 0; j < 16; ++j)
      tmp[j] = inb ? p[(size_t)j * HW] : (_Float16)0.f;
    _Float16* dst = &Bs[b][bN][bK16];
    *(v8h*)dst       = *(v8h*)&tmp[0];
    *(v8h*)(dst + 8) = *(v8h*)&tmp[8];
  };

  stageA(0, 0);
  stageB(0, 0);
  WAIT_ASYNC();
  __syncthreads();

  const int lm = lane & 15;
  const int hi = lane >> 4;

  for (int ks = 0; ks < NK; ++ks) {
    const int cur = ks & 1;
    if (ks + 2 < NK) __builtin_prefetch(aSrc + (size_t)(ks + 2) * 32, 0, 0);
    if (ks + 1 < NK) { stageA(ks + 1, cur ^ 1); stageB(ks + 1, cur ^ 1); }

    union F { v16h v; v8h h[2]; };
    F a[2], bb[4];
    #pragma unroll
    for (int mi = 0; mi < 2; ++mi) {
      const _Float16* ap = &As[cur][wm + mi * 16 + lm][hi * 8];
      a[mi].h[0] = *(const v8h*)ap;
      a[mi].h[1] = *(const v8h*)(ap + 16);
    }
    #pragma unroll
    for (int s = 0; s < 4; ++s) {
      const _Float16* bp = &Bs[cur][wn + s * 16 + lm][hi * 8];
      bb[s].h[0] = *(const v8h*)bp;
      bb[s].h[1] = *(const v8h*)(bp + 16);
    }
    #pragma unroll
    for (int mi = 0; mi < 2; ++mi)
      #pragma unroll
      for (int s = 0; s < 4; ++s)
        acc[mi][s] = __builtin_amdgcn_wmma_f32_16x16x32_f16(
            false, a[mi].v, false, bb[s].v, (short)0, acc[mi][s], false, false);

    WAIT_ASYNC();
    __syncthreads();
  }

  #pragma unroll
  for (int s = 0; s < 4; ++s) {
    const int p  = nBlk + wn + s * 16 + lm;
    const int ni = p / 784, px = p - ni * 784;
    float* ob = out + (size_t)ni * 512 * 784 + px;
    #pragma unroll
    for (int mi = 0; mi < 2; ++mi) {
      const int mBase = mBlk + wm + mi * 16 + hi * 8;
      #pragma unroll
      for (int v = 0; v < 8; ++v)
        ob[(size_t)(mBase + v) * 784] = acc[mi][s][v];
    }
  }
}


__global__ __launch_bounds__(256, 2) void conv2_iu8_k(
    const signed char* __restrict__ W,
    const unsigned char* __restrict__ act,
    const unsigned* __restrict__ mb,
    float* __restrict__ out)
{
  constexpr int CIN = 512, HIN = 28, HW = 784;
  constexpr int K = 9 * CIN;
  constexpr int NK = K / 64;

  __shared__ __align__(16) signed char   As[2][128][64];
  __shared__ __align__(16) unsigned char Bu[2][128][64];

  const int tid  = threadIdx.x;
  const int lane = tid & 31;
  const int wv   = tid >> 5;
  const int wm   = (wv >> 1) * 32;
  const int wn   = (wv & 1) * 64;
  const int mBlk = blockIdx.y * 128;
  const int nBlk = blockIdx.x * 128;

  const int aRow  = tid >> 1;
  const int aCol  = (tid & 1) * 32;
  const int bN    = tid >> 1;
  const int bK32  = (tid & 1) * 32;

  const int gp   = nBlk + bN;
  const int nImg = gp / 784;
  const int pix  = gp - nImg * 784;
  const int oy   = pix / 28;
  const int ox   = pix - oy * 28;

  const signed char*   aSrc = W + (size_t)(mBlk + aRow) * K + aCol;
  const unsigned char* actN = act + (size_t)nImg * CIN * HW;

  v8i acc[2][4] = {};

  auto stageA = [&](int ks, int b) {
    const signed char* src = aSrc + (size_t)ks * 64;
    signed char* dst = &As[b][aRow][aCol];
    ASYNC_B128(src, dst);
    ASYNC_B128(src + 16, dst + 16);
  };

  auto stageB = [&](int ks, int b) {
    const int k0 = ks * 64 + bK32;
    const int r  = k0 >> 9;
    const int c0 = k0 & 511;
    const int kh = r / 3, kw = r - kh * 3;
    const int iy = oy - 1 + kh, ix = ox - 1 + kw;
    const bool inb = (iy >= 0) && (iy < HIN) && (ix >= 0) && (ix < HIN);
    const unsigned char* p = actN + ((size_t)c0 * HIN + iy) * HIN + ix;
    union { v4i q[2]; unsigned char u[32]; } tmp;
    #pragma unroll
    for (int j = 0; j < 32; ++j)
      tmp.u[j] = inb ? p[(size_t)j * HW] : (unsigned char)0;
    unsigned char* dst = &Bu[b][bN][bK32];
    *(v4i*)dst        = tmp.q[0];
    *(v4i*)(dst + 16) = tmp.q[1];
  };

  stageA(0, 0);
  stageB(0, 0);
  WAIT_ASYNC();
  __syncthreads();

  const int lm = lane & 15;
  const int hi = lane >> 4;

  for (int ks = 0; ks < NK; ++ks) {
    const int cur = ks & 1;
    if (ks + 2 < NK) __builtin_prefetch(aSrc + (size_t)(ks + 2) * 64, 0, 0);
    if (ks + 1 < NK) { stageA(ks + 1, cur ^ 1); stageB(ks + 1, cur ^ 1); }

    union FA { v8i v; long long d[4]; };
    union FB { v8i v; v4i q[2]; long long d[4]; };
    FA a[2]; FB bb[4];
    #pragma unroll
    for (int mi = 0; mi < 2; ++mi) {
      const signed char* ap = &As[cur][wm + mi * 16 + lm][hi * 8];
      a[mi].d[0] = *(const long long*)(ap);
      a[mi].d[1] = *(const long long*)(ap + 16);
      a[mi].d[2] = *(const long long*)(ap + 32);
      a[mi].d[3] = *(const long long*)(ap + 48);
    }
    #pragma unroll
    for (int s = 0; s < 4; ++s) {
      const unsigned char* bp = &Bu[cur][wn + s * 16 + lm][hi * 8];
      bb[s].d[0] = *(const long long*)(bp);
      bb[s].d[1] = *(const long long*)(bp + 16);
      bb[s].d[2] = *(const long long*)(bp + 32);
      bb[s].d[3] = *(const long long*)(bp + 48);
    }
    #pragma unroll
    for (int mi = 0; mi < 2; ++mi)
      #pragma unroll
      for (int s = 0; s < 4; ++s)
        acc[mi][s] = __builtin_amdgcn_wmma_i32_16x16x64_iu8(
            true  , a[mi].v, false  , bb[s].v,
            acc[mi][s], false, false);

    WAIT_ASYNC();
    __syncthreads();
  }

  const float sc = __uint_as_float(*mb) * (1.f / 127.f) * (1.f / 255.f);
  #pragma unroll
  for (int s = 0; s < 4; ++s) {
    const int p  = nBlk + wn + s * 16 + lm;
    const int ni = p / 784, px = p - ni * 784;
    float* ob = out + (size_t)ni * 512 * 784 + px;
    #pragma unroll
    for (int mi = 0; mi < 2; ++mi) {
      const int mBase = mBlk + wm + mi * 16 + hi * 8;
      #pragma unroll
      for (int v = 0; v < 8; ++v)
        ob[(size_t)(mBase + v) * 784] = (float)acc[mi][s][v] * sc;
    }
  }
}


__global__ __launch_bounds__(256) void norm_quant_k(
    const float* __restrict__ h, const float* __restrict__ g, const float* __restrict__ b,
    unsigned char* __restrict__ o)
{
  const int nc = blockIdx.x;
  const int c  = nc & 511;
  const float* p = h + (size_t)nc * 784;
  const int t = threadIdx.x;
  float vals[4]; float s = 0.f, s2 = 0.f;
  #pragma unroll
  for (int i = 0; i < 4; ++i) {
    int idx = t + i * 256;
    float v = (idx < 784) ? p[idx] : 0.f;
    vals[i] = v; s += v; s2 += v * v;
  }
  #pragma unroll
  for (int off = 16; off; off >>= 1) { s += __shfl_xor(s, off, 32); s2 += __shfl_xor(s2, off, 32); }
  __shared__ float rs[8], rq[8];
  if ((t & 31) == 0) { rs[t >> 5] = s; rq[t >> 5] = s2; }
  __syncthreads();
  float S = 0.f, S2 = 0.f;
  #pragma unroll
  for (int w = 0; w < 8; ++w) { S += rs[w]; S2 += rq[w]; }
  const float mean = S * (1.f / 784.f);
  const float var  = S2 * (1.f / 784.f) - mean * mean;
  const float inv  = rsqrtf(var + 1e-5f);
  const float ga = g[c], be = b[c];
  #pragma unroll
  for (int i = 0; i < 4; ++i) {
    int idx = t + i * 256;
    if (idx < 784) {
      float y = (vals[i] - mean) * inv * ga + be;
      y = fminf(fmaxf(y, 0.f), 1.f);
      int k = (int)rintf(y * 255.f);
      o[(size_t)nc * 784 + idx] = (unsigned char)k;
    }
  }
}


__global__ __launch_bounds__(256) void norm_add_k(
    const float* __restrict__ h2, const float* __restrict__ sr,
    const float* __restrict__ g2, const float* __restrict__ b2,
    const float* __restrict__ gp, const float* __restrict__ bp,
    float* __restrict__ out)
{
  const int nc = blockIdx.x;
  const int c  = nc & 511;
  const float* ph = h2 + (size_t)nc * 784;
  const float* ps = sr + (size_t)nc * 784;
  const int t = threadIdx.x;
  float hv[4], sv[4];
  float s1 = 0.f, q1 = 0.f, s2 = 0.f, q2 = 0.f;
  #pragma unroll
  for (int i = 0; i < 4; ++i) {
    int idx = t + i * 256;
    float a = (idx < 784) ? ph[idx] : 0.f;
    float d = (idx < 784) ? ps[idx] : 0.f;
    hv[i] = a; sv[i] = d;
    s1 += a; q1 += a * a; s2 += d; q2 += d * d;
  }
  #pragma unroll
  for (int off = 16; off; off >>= 1) {
    s1 += __shfl_xor(s1, off, 32); q1 += __shfl_xor(q1, off, 32);
    s2 += __shfl_xor(s2, off, 32); q2 += __shfl_xor(q2, off, 32);
  }
  __shared__ float r0[8], r1[8], r2[8], r3[8];
  if ((t & 31) == 0) { int w = t >> 5; r0[w] = s1; r1[w] = q1; r2[w] = s2; r3[w] = q2; }
  __syncthreads();
  float A = 0.f, B = 0.f, C = 0.f, D = 0.f;
  #pragma unroll
  for (int w = 0; w < 8; ++w) { A += r0[w]; B += r1[w]; C += r2[w]; D += r3[w]; }
  const float mh = A * (1.f / 784.f);
  const float ih = rsqrtf(B * (1.f / 784.f) - mh * mh + 1e-5f);
  const float ms = C * (1.f / 784.f);
  const float is = rsqrtf(D * (1.f / 784.f) - ms * ms + 1e-5f);
  const float ga2 = g2[c], be2 = b2[c], gap = gp[c], bep = bp[c];
  #pragma unroll
  for (int i = 0; i < 4; ++i) {
    int idx = t + i * 256;
    if (idx < 784) {
      float y = (hv[i] - mh) * ih * ga2 + be2 + (sv[i] - ms) * is * gap + bep;
      out[(size_t)nc * 784 + idx] = y;
    }
  }
}


extern "C" void kernel_launch(void* const* d_in, const int* in_sizes, int n_in,
                              void* d_out, int out_size, void* d_ws, size_t ws_size,
                              hipStream_t stream)
{
  (void)in_sizes; (void)n_in; (void)out_size; (void)ws_size;
  const float* x  = (const float*)d_in[0];
  const float* w1 = (const float*)d_in[1];
  const float* g1 = (const float*)d_in[2];
  const float* b1 = (const float*)d_in[3];
  const float* w2 = (const float*)d_in[4];
  const float* g2 = (const float*)d_in[5];
  const float* b2 = (const float*)d_in[6];
  const float* wp = (const float*)d_in[7];
  const float* gp = (const float*)d_in[8];
  const float* bp = (const float*)d_in[9];
  float* out = (float*)d_out;

  char* ws = (char*)d_ws;
  size_t off = 0;
  auto take = [&](size_t bytes) { char* p = ws + off; off = (off + bytes + 255) & ~(size_t)255; return p; };

  const int NX   = 32 * 256 * 56 * 56;
  const int NOUT = 32 * 512 * 28 * 28;

  _Float16*      xh   = (_Float16*)take((size_t)NX * 2);
  _Float16*      w1q  = (_Float16*)take((size_t)512 * 2304 * 2);
  signed char*   w2q8 = (signed char*)take((size_t)512 * 4608);
  _Float16*      wpq  = (_Float16*)take((size_t)512 * 256 * 2);
  float*         h1   = (float*)take((size_t)NOUT * 4);
  unsigned char* a2u  = (unsigned char*)take((size_t)NOUT);
  float*         h2   = (float*)take((size_t)NOUT * 4);
  float*         sr   = (float*)take((size_t)NOUT * 4);
  unsigned*      scl  = (unsigned*)take(3 * 4);

  zero_u32_k<<<1, 32, 0, stream>>>(scl, 3);
  maxabs_k<<<512, 256, 0, stream>>>(w1, 512 * 256 * 9, scl + 0);
  maxabs_k<<<512, 256, 0, stream>>>(w2, 512 * 512 * 9, scl + 1);
  maxabs_k<<<256, 256, 0, stream>>>(wp, 512 * 256,     scl + 2);

  cvt_f16_k<<<2048, 256, 0, stream>>>(x, xh, NX);

  pack_w_k<256, 3><<<1024, 256, 0, stream>>>(w1, scl + 0, w1q, 512);
  pack_w8_k<512, 3><<<2048, 256, 0, stream>>>(w2, scl + 1, w2q8, 512);
  pack_w_k<256, 1><<<256,  256, 0, stream>>>(wp, scl + 2, wpq, 512);

  dim3 cgrid(196, 4);
  conv_wmma_k<256, 3, 2, 1, 56><<<cgrid, 256, 0, stream>>>(w1q, xh, h1);
  norm_quant_k<<<16384, 256, 0, stream>>>(h1, g1, b1, a2u);
  conv2_iu8_k<<<cgrid, 256, 0, stream>>>(w2q8, a2u, scl + 1, h2);
  conv_wmma_k<256, 1, 2, 0, 56><<<cgrid, 256, 0, stream>>>(wpq, xh, sr);
  norm_add_k<<<16384, 256, 0, stream>>>(h2, sr, g2, b2, gp, bp, out);
}
